// MultiHeadSelfAttention_24232205484570
// MI455X (gfx1250) — matrix operand order repaired and checked by reading, not compiled, not run on hardware
//
#include <hip/hip_runtime.h>
#include <hip/hip_bf16.h>
#include <math.h>

// MHA on MI455X (gfx1250): all GEMMs via v_wmma_f32_16x16x32_bf16 (wave32),
// flash-style attention (no S x S materialization), bf16 data / fp32 accum.
// K-loops use unroll-by-2 ping-pong register buffering: fragments for the
// next K-chunk are in flight behind the current chunk's WMMAs, with no
// register-rotation mov chains.

typedef __attribute__((ext_vector_type(16))) __bf16 v16bf;
typedef __attribute__((ext_vector_type(8)))  __bf16 v8bf;
typedef __attribute__((ext_vector_type(4)))  __bf16 v4bf;
typedef __attribute__((ext_vector_type(8)))  float  v8f;

#define D_MODEL 1024
#define SEQ     2048
#define NBATCH  2
#define NHEADS  16
#define DK      64
#define MROWS   (NBATCH * SEQ)   // 4096

// ---------------------------------------------------------------------------
// WMMA helpers (CDNA5 documented VGPR layouts, 16-bit data)
// A 16x32 (MxK): lane L -> m=L%16, h=L/16; elem e<8 -> k=kbase+8h+e,
//                e>=8 -> k=kbase+8h+8+e  (two contiguous 8-elem runs)
// B 32x16 (KxN): lane L -> n=L%16, h=L/16; elem e -> k as for A (two 8-elem runs)
// C/D 16x16 f32: lane L -> n=L%16; VGPR v -> m = v + 8*(L/16)
// ---------------------------------------------------------------------------

__device__ __forceinline__ v8f wmma_bf16(v16bf a, v16bf b, v8f c) {
  // (neg_a, A, neg_b, B, c_mod, C, reuse_a, reuse_b)
  return __builtin_amdgcn_wmma_f32_16x16x32_bf16(false, a, false, b,
                                                 (short)0, c, false, false);
}

__device__ __forceinline__ v16bf load_afrag(const __bf16* __restrict__ base,
                                            int row0, int ld, int kbase, int lane) {
  int m = lane & 15, h = lane >> 4;
  const __bf16* p = base + (size_t)(row0 + m) * ld + kbase + 8 * h;
  v8bf lo = *(const v8bf*)(p);        // k = kbase+8h .. +7
  v8bf hi = *(const v8bf*)(p + 16);   // k = kbase+8h+16 .. +23
  v16bf r;
#pragma unroll
  for (int e = 0; e < 8; ++e) { r[e] = lo[e]; r[e + 8] = hi[e]; }
  return r;
}

// B fragment from row-major [N x K] storage (column n of B == row n of source)
__device__ __forceinline__ v16bf load_bfrag(const __bf16* __restrict__ base,
                                            int row0, int ld, int kbase, int lane) {
  int n = lane & 15, h = lane >> 4;
  const __bf16* p = base + (size_t)(row0 + n) * ld + kbase + 8 * h;
  v8bf lo = *(const v8bf*)(p);
  v8bf hi = *(const v8bf*)(p + 16);
  v16bf r;
#pragma unroll
  for (int e = 0; e < 8; ++e) { r[e] = lo[e]; r[e + 8] = hi[e]; }
  return r;
}

// ---------------------------------------------------------------------------
// fp32 -> bf16 conversion (vectorized x4)
// ---------------------------------------------------------------------------
__global__ void __launch_bounds__(256) cvt_f32_bf16_kernel(
    const float4* __restrict__ in, __bf16* __restrict__ out, int n4) {
  int i = blockIdx.x * blockDim.x + threadIdx.x;
  if (i < n4) {
    float4 f = in[i];
    v4bf r;
    r[0] = (__bf16)f.x; r[1] = (__bf16)f.y; r[2] = (__bf16)f.z; r[3] = (__bf16)f.w;
    *(v4bf*)(out + 4 * (size_t)i) = r;
  }
}

// ---------------------------------------------------------------------------
// Projection GEMM: D[m,n] = sum_k X[m,k] * W[n,k] + bias[n]
// X: bf16 [4096 x 1024] row-major; W: bf16 [1024 x 1024] row-major ([out,in])
// block = 128 threads = 4 waves; each wave: 16x64 tile. grid = (M/64, N/64).
// Unrolled-by-2 ping-pong buffering over K (no rotation copies).
// mode 0: fp32 out [m,n]            (final output projection)
// mode 1: bf16 -> Qh  [B,H,S,dk]    (scale = 1/sqrt(dk) folded in)
// mode 2: bf16 -> Kh  [B,H,S,dk]
// mode 3: bf16 -> VhT [B,H,dk,S]    (transposed for contiguous PV B-columns)
// ---------------------------------------------------------------------------
__global__ void __launch_bounds__(128) gemm_proj_kernel(
    const __bf16* __restrict__ X, const __bf16* __restrict__ W,
    const float* __restrict__ bias, float* __restrict__ outf,
    __bf16* __restrict__ outbf, int mode, float scale) {
  int lane = threadIdx.x & 31;
  int wave = threadIdx.x >> 5;
  int m0 = (blockIdx.x * 4 + wave) * 16;
  int n0 = blockIdx.y * 64;

  v8f acc[4] = {};

  // buffer set A: fragments for k = 0
  v16bf aA = load_afrag(X, m0, D_MODEL, 0, lane);
  v16bf bA[4];
#pragma unroll
  for (int t = 0; t < 4; ++t) bA[t] = load_bfrag(W, n0 + 16 * t, D_MODEL, 0, lane);
  v16bf aB, bB[4];

  for (int kb = 0; kb < D_MODEL; kb += 64) {
    int k1 = kb + 32;
    // issue loads for k1 into buffer set B (fly behind the kb WMMAs)
    __builtin_prefetch(X + (size_t)(m0 + (lane & 15)) * D_MODEL + k1 + 32, 0, 1);
    aB = load_afrag(X, m0, D_MODEL, k1, lane);
#pragma unroll
    for (int t = 0; t < 4; ++t) bB[t] = load_bfrag(W, n0 + 16 * t, D_MODEL, k1, lane);
    // compute chunk kb from buffer set A
#pragma unroll
    for (int t = 0; t < 4; ++t) acc[t] = wmma_bf16(aA, bA[t], acc[t]);

    // issue loads for k2 into buffer set A (wraps to 0 on last iter; those
    // results are never consumed — keeps the loop branch-free and in-bounds)
    int k2 = (kb + 64) & (D_MODEL - 1);
    aA = load_afrag(X, m0, D_MODEL, k2, lane);
#pragma unroll
    for (int t = 0; t < 4; ++t) bA[t] = load_bfrag(W, n0 + 16 * t, D_MODEL, k2, lane);
    // compute chunk k1 from buffer set B
#pragma unroll
    for (int t = 0; t < 4; ++t) acc[t] = wmma_bf16(aB, bB[t], acc[t]);
  }

  int n = lane & 15, h = lane >> 4;
#pragma unroll
  for (int t = 0; t < 4; ++t) {
#pragma unroll
    for (int v = 0; v < 8; ++v) {
      int gm = m0 + v + 8 * h;          // row in [0, 4096)
      int gn = n0 + 16 * t + n;         // col in [0, 1024)
      float val = acc[t][v] + bias[gn];
      if (mode == 0) {
        outf[(size_t)gm * D_MODEL + gn] = val;
      } else {
        __bf16 bv_ = (__bf16)(val * scale);
        int b    = gm >> 11;            // / SEQ
        int s    = gm & (SEQ - 1);
        int head = gn >> 6;             // / DK
        int d    = gn & (DK - 1);
        if (mode == 3)
          outbf[(((size_t)(b * NHEADS + head)) * DK + d) * SEQ + s] = bv_;
        else
          outbf[(((size_t)(b * NHEADS + head)) * SEQ + s) * DK + d] = bv_;
      }
    }
  }
}

// ---------------------------------------------------------------------------
// Flash attention: per wave one 16-query tile of one (b,h); stream keys in
// chunks of 32 with online softmax. Score scale pre-folded into Qh.
// V fragments (this chunk) and K fragments (next chunk, wrap-indexed) are
// issued right after the score WMMAs so they fly across the softmax VALU
// work and the LDS transpose round-trip.
// Qh,Kh: [B*H, S, dk] bf16 ; VhT: [B*H, dk, S] bf16 ; AO: [B, S, H*dk] bf16
// ---------------------------------------------------------------------------
__global__ void __launch_bounds__(128) flash_attn_kernel(
    const __bf16* __restrict__ Qh, const __bf16* __restrict__ Kh,
    const __bf16* __restrict__ VhT, __bf16* __restrict__ AO) {
  __shared__ __bf16 lds_p[4][16][32];   // per-wave 16x32 P tile (D->A transpose)

  int lane = threadIdx.x & 31;
  int wave = threadIdx.x >> 5;
  int tile = blockIdx.x * 4 + wave;     // 4096 query tiles total
  int bh   = tile >> 7;                 // 128 tiles of 16 queries per (b,h)
  int q0   = (tile & 127) * 16;

  const __bf16* Qb = Qh  + (size_t)bh * SEQ * DK;
  const __bf16* Kb = Kh  + (size_t)bh * SEQ * DK;
  const __bf16* Vb = VhT + (size_t)bh * DK * SEQ;

  // Q fragments for the full dk=64 (two K-chunks of 32), reused every step
  v16bf qf0 = load_afrag(Qb, q0, DK, 0,  lane);
  v16bf qf1 = load_afrag(Qb, q0, DK, 32, lane);

  v8f acc[4] = {};
  float mrow[8], lrow[8];
#pragma unroll
  for (int v = 0; v < 8; ++v) { mrow[v] = -__builtin_inff(); lrow[v] = 0.0f; }

  int m = lane & 15, h = lane >> 4;

  // prologue: K fragments for kv = 0
  v16bf kf0 = load_bfrag(Kb, 0,  DK, 0,  lane);
  v16bf kf1 = load_bfrag(Kb, 0,  DK, 32, lane);
  v16bf kf2 = load_bfrag(Kb, 16, DK, 0,  lane);
  v16bf kf3 = load_bfrag(Kb, 16, DK, 32, lane);

  for (int kv = 0; kv < SEQ; kv += 32) {
    // ---- scores S = Q (16x64) x K^T (64x32), two 16x16 D tiles ----
    v8f c0 = {}, c1 = {};
    c0 = wmma_bf16(qf0, kf0, c0);
    c0 = wmma_bf16(qf1, kf1, c0);
    c1 = wmma_bf16(qf0, kf2, c1);
    c1 = wmma_bf16(qf1, kf3, c1);

    // ---- issue V loads (this chunk) + K loads (next chunk, wrap on last
    //      iteration: in-bounds, results unused) ----
    v16bf vb[4];
#pragma unroll
    for (int t = 0; t < 4; ++t) vb[t] = load_bfrag(Vb, 16 * t, SEQ, kv, lane);
    int kvn = (kv + 32) & (SEQ - 1);
    kf0 = load_bfrag(Kb, kvn,      DK, 0,  lane);
    kf1 = load_bfrag(Kb, kvn,      DK, 32, lane);
    kf2 = load_bfrag(Kb, kvn + 16, DK, 0,  lane);
    kf3 = load_bfrag(Kb, kvn + 16, DK, 32, lane);

    // ---- online softmax, rows distributed as (v + 8h) across half-waves ----
#pragma unroll
    for (int v = 0; v < 8; ++v) {
      float s = fmaxf(c0[v], c1[v]);
      s = fmaxf(s, __shfl_xor(s, 1, 32));
      s = fmaxf(s, __shfl_xor(s, 2, 32));
      s = fmaxf(s, __shfl_xor(s, 4, 32));
      s = fmaxf(s, __shfl_xor(s, 8, 32));
      float mn = fmaxf(mrow[v], s);
      float r  = __expf(mrow[v] - mn);
      float p0 = __expf(c0[v] - mn);
      float p1 = __expf(c1[v] - mn);
      float ps = p0 + p1;
      ps += __shfl_xor(ps, 1, 32);
      ps += __shfl_xor(ps, 2, 32);
      ps += __shfl_xor(ps, 4, 32);
      ps += __shfl_xor(ps, 8, 32);
      lrow[v] = lrow[v] * r + ps;
      mrow[v] = mn;
#pragma unroll
      for (int t = 0; t < 4; ++t) acc[t][v] *= r;
      int row = v + 8 * h;
      lds_p[wave][row][m]      = (__bf16)p0;   // cols kv..kv+15
      lds_p[wave][row][m + 16] = (__bf16)p1;   // cols kv+16..kv+31
    }
    __syncthreads();   // uniform trip count for all 4 waves

    // ---- P tile back as A fragment (LDS transpose) ----
    v16bf pa;
#pragma unroll
    for (int e = 0; e < 8; ++e) {
      pa[e]     = lds_p[wave][m][8 * h + e];
      pa[e + 8] = lds_p[wave][m][8 * h + 16 + e];
    }

    // ---- acc += P (16x32) x V (32x64), 4 dk-subtiles ----
#pragma unroll
    for (int t = 0; t < 4; ++t) acc[t] = wmma_bf16(pa, vb[t], acc[t]);
    __syncthreads();
  }

  // ---- epilogue: normalize and scatter to [B, S, H*dk] ----
  int b = bh >> 4, head = bh & 15;
#pragma unroll
  for (int t = 0; t < 4; ++t) {
#pragma unroll
    for (int v = 0; v < 8; ++v) {
      int s = q0 + v + 8 * h;
      int d = head * DK + 16 * t + m;
      AO[((size_t)(b * SEQ + s)) * D_MODEL + d] = (__bf16)(acc[t][v] / lrow[v]);
    }
  }
}

// ---------------------------------------------------------------------------
// Host launcher. Workspace layout (64 MB):
//   Xq/Xk/Xv bf16 (8MB each), Wq/Wk/Wv/Wo bf16 (2MB each),
//   Qh/Kh (8MB each), VhT (8MB), AO (8MB)
// ---------------------------------------------------------------------------
extern "C" void kernel_launch(void* const* d_in, const int* in_sizes, int n_in,
                              void* d_out, int out_size, void* d_ws, size_t ws_size,
                              hipStream_t stream) {
  const float* q  = (const float*)d_in[0];
  const float* k  = (const float*)d_in[1];
  const float* v  = (const float*)d_in[2];
  const float* Wq = (const float*)d_in[3];
  const float* bq = (const float*)d_in[4];
  const float* Wk = (const float*)d_in[5];
  const float* bk = (const float*)d_in[6];
  const float* Wv = (const float*)d_in[7];
  const float* bv = (const float*)d_in[8];
  const float* Wo = (const float*)d_in[9];
  const float* bo = (const float*)d_in[10];

  char* ws = (char*)d_ws;
  const size_t MB = 1u << 20;
  __bf16* Xq  = (__bf16*)(ws + 0 * MB);
  __bf16* Xk  = (__bf16*)(ws + 8 * MB);
  __bf16* Xv  = (__bf16*)(ws + 16 * MB);
  __bf16* Wqb = (__bf16*)(ws + 24 * MB);
  __bf16* Wkb = (__bf16*)(ws + 26 * MB);
  __bf16* Wvb = (__bf16*)(ws + 28 * MB);
  __bf16* Wob = (__bf16*)(ws + 30 * MB);
  __bf16* Qh  = (__bf16*)(ws + 32 * MB);
  __bf16* Kh  = (__bf16*)(ws + 40 * MB);
  __bf16* VhT = (__bf16*)(ws + 48 * MB);
  __bf16* AO  = (__bf16*)(ws + 56 * MB);

  const int nX = MROWS * D_MODEL;        // 4096*1024
  const int nW = D_MODEL * D_MODEL;      // 1024*1024

  cvt_f32_bf16_kernel<<<nX / 4 / 256, 256, 0, stream>>>((const float4*)q,  Xq,  nX / 4);
  cvt_f32_bf16_kernel<<<nX / 4 / 256, 256, 0, stream>>>((const float4*)k,  Xk,  nX / 4);
  cvt_f32_bf16_kernel<<<nX / 4 / 256, 256, 0, stream>>>((const float4*)v,  Xv,  nX / 4);
  cvt_f32_bf16_kernel<<<nW / 4 / 256, 256, 0, stream>>>((const float4*)Wq, Wqb, nW / 4);
  cvt_f32_bf16_kernel<<<nW / 4 / 256, 256, 0, stream>>>((const float4*)Wk, Wkb, nW / 4);
  cvt_f32_bf16_kernel<<<nW / 4 / 256, 256, 0, stream>>>((const float4*)Wv, Wvb, nW / 4);
  cvt_f32_bf16_kernel<<<nW / 4 / 256, 256, 0, stream>>>((const float4*)Wo, Wob, nW / 4);

  dim3 gp(MROWS / 64, D_MODEL / 64);     // (64, 16)
  gemm_proj_kernel<<<gp, 128, 0, stream>>>(Xq, Wqb, bq, nullptr, Qh,  1, 0.125f);
  gemm_proj_kernel<<<gp, 128, 0, stream>>>(Xk, Wkb, bk, nullptr, Kh,  2, 1.0f);
  gemm_proj_kernel<<<gp, 128, 0, stream>>>(Xv, Wvb, bv, nullptr, VhT, 3, 1.0f);

  flash_attn_kernel<<<(NBATCH * NHEADS * (SEQ / 16)) / 4, 128, 0, stream>>>(Qh, Kh, VhT, AO);

  gemm_proj_kernel<<<gp, 128, 0, stream>>>(AO, Wob, bo, (float*)d_out, nullptr, 0, 1.0f);
}
